// AttentionFrame_52853867545005
// MI455X (gfx1250) — matrix operand order repaired and checked by reading, not compiled, not run on hardware
//
#include <hip/hip_runtime.h>
#include <hip/hip_bf16.h>
#include <math.h>
#include <stdint.h>

// ---------------------------------------------------------------------------
// MI455X (gfx1250) implementation.
// Roofline: ~110MB of unavoidable traffic @ 23.3TB/s dominates; total math is
// ~5.4 GFLOP. Plan:
//  - GEMM1..3 : f16 WMMA 16x16x32 (f32 accum), B-panels double-buffered into
//               LDS with global_load_async_to_lds_b128 + s_wait_asynccnt.
//  - Wa GEMM  : f32 WMMA 16x16x4 streaming Wa (64MB) row-major exactly once;
//               K-split partials combined with global_atomic_add_f32.
// ---------------------------------------------------------------------------

typedef __attribute__((ext_vector_type(16))) _Float16 v16h;
typedef __attribute__((ext_vector_type(8)))  _Float16 v8h;
typedef __attribute__((ext_vector_type(8)))  float    v8f;
typedef __attribute__((ext_vector_type(2)))  float    v2f;
typedef int v4i_gcc __attribute__((vector_size(16)));   // matches builtin param

#define ALPHA1 0.9f
#define ALPHA2 1.35f          /* ALPHA * 1.5 */
#define LN_EPS 1e-3f

static constexpr int Bb = 32, Tt = 256, Dd = 512;
static constexpr int M1 = Bb * Tt;           // 8192 rows for GEMM1..3
static constexpr int NN = Tt * Tt;           // 65536 (flattened feature dim)

union AFrag { v16h v; v8h h[2]; };

// ---- CDNA5 async global->LDS copy (ASYNCcnt-tracked), with sync fallback --
#if __has_builtin(__builtin_amdgcn_global_load_async_to_lds_b128)
#define HAVE_ASYNC_LDS 1
#else
#define HAVE_ASYNC_LDS 0
#endif

#if HAVE_ASYNC_LDS
#if __has_builtin(__builtin_amdgcn_s_wait_asynccnt)
#define ASYNC_WAIT(n) __builtin_amdgcn_s_wait_asynccnt(n)
#else
#define ASYNC_WAIT(n) asm volatile("s_wait_asynccnt %0" ::"i"(n) : "memory")
#endif
#else
#define ASYNC_WAIT(n) ((void)0)
#endif

__device__ __forceinline__ void copy16_to_lds(const _Float16* g, _Float16* l) {
#if HAVE_ASYNC_LDS
    __builtin_amdgcn_global_load_async_to_lds_b128(
        (__attribute__((address_space(1))) v4i_gcc*)(uintptr_t)g,
        (__attribute__((address_space(3))) v4i_gcc*)(uint32_t)(uintptr_t)l,
        0, 0);
#else
    *(v8h*)l = *(const v8h*)g;
#endif
}

// ---------------- EMA: two chained scans fused, one thread per (b,d) -------
__global__ void ema_kernel(const float* __restrict__ x, _Float16* __restrict__ e) {
    int idx = blockIdx.x * blockDim.x + threadIdx.x;     // 0 .. B*D-1
    int b = idx / Dd, d = idx % Dd;
    const float* xp = x + (size_t)b * Tt * Dd + d;
    _Float16*    ep = e + (size_t)b * Tt * Dd + d;
    float c1 = xp[0];          // ema1[0] = x[0]
    float c2 = c1;             // ema2[0] = ema1[0]
    ep[0] = (_Float16)(c1 + c2);
    for (int t = 1; t < Tt; ++t) {
        float xv = xp[(size_t)t * Dd];
        c1 = ALPHA1 * xv + (1.0f - ALPHA1) * c1;
        c2 = ALPHA2 * c1 + (1.0f - ALPHA2) * c2;
        ep[(size_t)t * Dd] = (_Float16)(c1 + c2);
    }
}

// ---------------- weight transpose + f32->f16 convert: Wt[n*K+k]=W[k*N+n] --
__global__ void wprep_kernel(const float* __restrict__ W, _Float16* __restrict__ Wt,
                             int K, int N) {
    int idx = blockIdx.x * blockDim.x + threadIdx.x;
    if (idx >= K * N) return;
    int k = idx / N, n = idx % N;
    Wt[(size_t)n * K + k] = (_Float16)W[idx];
}

// ---------------- f16 WMMA GEMM with async-LDS double-buffered B panels ----
// Block = 8 waves = 2 m-tiles x 4 n-groups; block stages the full 256-col
// B k-panel (256x32 f16 = 16KB) into LDS, double buffered via ASYNCcnt.
// A: [M,K] f16 row-major. Bt: [256,K] f16 (pre-transposed).
// ACT: 0 tanh->f16, 1 relu->f16, 2 identity->f32.
template <int ACT>
__global__ void gemm16_kernel(const _Float16* __restrict__ A,
                              const _Float16* __restrict__ Bt,
                              const float* __restrict__ bias,
                              _Float16* __restrict__ Oh,
                              float* __restrict__ Of,
                              int M, int N, int K) {
    __shared__ alignas(16) _Float16 panel[2][256 * 32];   // 2 x 16KB

    const int tid    = threadIdx.x;
    const int lane   = tid & 31;
    const int w      = __builtin_amdgcn_readfirstlane(tid >> 5);   // scalar wave id
    const int ngroup = w & 3;
    const int mtile  = blockIdx.x * 2 + (w >> 2);
    const int r   = lane & 15;
    const int kA  = (lane < 16) ? 0 : 8;   // A: halves 0-7 -> K=kA.., 8-15 -> K=kA+16..
    const int kB  = (lane < 16) ? 0 : 8;   // B: halves 0-7 -> K=kB.., 8-15 -> K=kB+16..
    const _Float16* arow = A + (size_t)(mtile * 16 + r) * K;

    // stage one 256x32 B k-panel: 1024 x 16B chunks, 4 per thread per wave
    auto issue_panel = [&](int k0, int buf) {
#pragma unroll
        for (int it = 0; it < 4; ++it) {
            int c = tid + it * 256;          // chunk id
            int n = c >> 2, part = c & 3;    // 4 x 16B per 64B row
            copy16_to_lds(Bt + (size_t)n * K + k0 + part * 8,
                          &panel[buf][n * 32 + part * 8]);
        }
    };

    const int nsteps = K >> 5;
    v8f acc[4] = {};
    issue_panel(0, 0);
    for (int s = 0; s < nsteps; ++s) {
        const int k0 = s << 5;
        if (s + 1 < nsteps) { issue_panel(k0 + 32, (s + 1) & 1); ASYNC_WAIT(4); }
        else                { ASYNC_WAIT(0); }
        __syncthreads();                      // panel[s&1] complete across waves

        AFrag a;
        a.h[0] = *(const v8h*)(arow + k0 + kA);
        a.h[1] = *(const v8h*)(arow + k0 + kA + 16);
        const _Float16* pb = &panel[s & 1][0];
#pragma unroll
        for (int j = 0; j < 4; ++j) {
            const _Float16* bp = pb + ((ngroup * 4 + j) * 16 + r) * 32 + kB;
            AFrag bf;
            bf.h[0] = *(const v8h*)(bp);      // ds_load_b128
            bf.h[1] = *(const v8h*)(bp + 16);
            acc[j] = __builtin_amdgcn_wmma_f32_16x16x32_f16(
                false, a.v, false, bf.v, (short)0, acc[j], false, false);
        }
        __syncthreads();                      // done reading before overwrite
    }

    // C/D layout: VGPR v holds (M = v + 8*(lane/16), N = lane%16)
    const int mbase = mtile * 16 + ((lane >> 4) << 3);
#pragma unroll
    for (int j = 0; j < 4; ++j) {
        int n = (ngroup * 4 + j) * 16 + r;
        float bv = bias[n];
#pragma unroll
        for (int v = 0; v < 8; ++v) {
            float val = acc[j][v] + bv;
            if (ACT == 0) val = tanhf(val);
            if (ACT == 1) val = fmaxf(val, 0.0f);
            size_t off = (size_t)(mbase + v) * N + n;
            if (ACT == 2) Of[off] = val;
            else          Oh[off] = (_Float16)val;
        }
    }
}

// ---------------- LayerNorm over 65536 per batch row -----------------------
__global__ void ln_kernel(const float* __restrict__ f,
                          const float* __restrict__ gamma,
                          const float* __restrict__ beta,
                          float* __restrict__ fn) {
    __shared__ float s1[256], s2[256];
    const int b = blockIdx.x, tid = threadIdx.x;
    const float* row = f + (size_t)b * NN;
    float s = 0.f, sq = 0.f;
    for (int i = tid; i < NN; i += 256) { float v = row[i]; s += v; sq += v * v; }
    s1[tid] = s; s2[tid] = sq;
    __syncthreads();
    for (int st = 128; st > 0; st >>= 1) {
        if (tid < st) { s1[tid] += s1[tid + st]; s2[tid] += s2[tid + st]; }
        __syncthreads();
    }
    float mu  = s1[0] * (1.0f / NN);
    float var = s2[0] * (1.0f / NN) - mu * mu;
    float inv = rsqrtf(var + LN_EPS);
    float* orow = fn + (size_t)b * NN;
    for (int i = tid; i < NN; i += 256)
        orow[i] = (row[i] - mu) * inv * gamma[i] + beta[i];
}

__global__ void zero_kernel(float* __restrict__ p, int n) {
    int i = blockIdx.x * blockDim.x + threadIdx.x;
    if (i < n) p[i] = 0.0f;
}

// ---------------- f32 WMMA GEMM: g_pre += fn[32,65536] @ Wa[65536,256] -----
// Wave owns (n-group of 4 tiles) x (K-slice of 256); both M-tiles in-wave so
// the 64MB Wa stream is read exactly once. Wave-uniform indices forced scalar
// via readfirstlane so the K loop is a scalar s_cbranch loop (EXEC stays all
// ones across the WMMAs). Partials combined with global_atomic_add_f32.
__global__ void gemmwa_kernel(const float* __restrict__ Afn,
                              const float* __restrict__ Wa,
                              float* __restrict__ gpre) {
    const int lane   = threadIdx.x & 31;
    const int wid    = __builtin_amdgcn_readfirstlane(
                           blockIdx.x * (blockDim.x >> 5) + (threadIdx.x >> 5));
    const int ngroup = wid & 3;
    const int kslice = wid >> 2;                 // 0..255
    const int kbeg = kslice * (NN / 256);
    const int kend = kbeg + (NN / 256);
    const int ra = lane & 15;
    const int ka = (lane < 16) ? 0 : 2;          // f32 A 16x4: V0=K(ka), V1=K(ka+1)
    const int nb = lane & 15;
    const int kb = (lane < 16) ? 0 : 2;          // f32 B 4x16 mirrors A
    const float* a0p = Afn + (size_t)ra * NN;
    const float* a1p = Afn + (size_t)(16 + ra) * NN;

    v8f acc[2][4] = {};
    for (int k0 = kbeg; k0 < kend; k0 += 4) {
        v2f a0 = *(const v2f*)(a0p + k0 + ka);
        v2f a1 = *(const v2f*)(a1p + k0 + ka);
#pragma unroll
        for (int j = 0; j < 4; ++j) {
            int n = (ngroup * 4 + j) * 16 + nb;
            v2f bf;
            bf.x = Wa[(size_t)(k0 + kb) * Tt + n];
            bf.y = Wa[(size_t)(k0 + kb + 1) * Tt + n];
            acc[0][j] = __builtin_amdgcn_wmma_f32_16x16x4_f32(
                false, a0, false, bf, (short)0, acc[0][j], false, false);
            acc[1][j] = __builtin_amdgcn_wmma_f32_16x16x4_f32(
                false, a1, false, bf, (short)0, acc[1][j], false, false);
        }
    }
    const int mb = (lane >> 4) << 3;
#pragma unroll
    for (int mt = 0; mt < 2; ++mt)
#pragma unroll
        for (int j = 0; j < 4; ++j) {
            int n = (ngroup * 4 + j) * 16 + nb;
#pragma unroll
            for (int v = 0; v < 8; ++v)
                atomicAdd(&gpre[(size_t)(mt * 16 + mb + v) * Tt + n], acc[mt][j][v]);
        }
}

// ---------------- finalize: out = sigmoid(g_pre + ba) * x ------------------
__global__ void finalize_kernel(const float* __restrict__ x,
                                const float* __restrict__ gpre,
                                const float* __restrict__ ba,
                                float* __restrict__ out) {
    size_t idx = (size_t)blockIdx.x * blockDim.x + threadIdx.x;
    size_t bt = idx / Dd;
    int t = (int)(bt % Tt);
    float gp = gpre[bt] + ba[t];
    float gate = 1.0f / (1.0f + expf(-gp));
    out[idx] = gate * x[idx];
}

// ---------------------------------------------------------------------------
extern "C" void kernel_launch(void* const* d_in, const int* in_sizes, int n_in,
                              void* d_out, int out_size, void* d_ws, size_t ws_size,
                              hipStream_t stream) {
    const float* x     = (const float*)d_in[0];
    const float* W1    = (const float*)d_in[1];
    const float* b1    = (const float*)d_in[2];
    const float* W2    = (const float*)d_in[3];
    const float* b2    = (const float*)d_in[4];
    const float* W3    = (const float*)d_in[5];
    const float* b3    = (const float*)d_in[6];
    const float* gamma = (const float*)d_in[7];
    const float* beta  = (const float*)d_in[8];
    const float* Wa    = (const float*)d_in[9];
    const float* ba    = (const float*)d_in[10];
    float* out = (float*)d_out;

    // workspace carve (~33 MB total)
    size_t off = 0;
    auto carve = [&](size_t bytes) {
        void* p = (char*)d_ws + off;
        off += (bytes + 255) & ~(size_t)255;
        return p;
    };
    _Float16* e_h  = (_Float16*)carve((size_t)Bb * Tt * Dd * 2);  // 8 MB
    _Float16* h1   = (_Float16*)carve((size_t)M1 * Tt * 2);       // 4 MB
    _Float16* h2   = (_Float16*)carve((size_t)M1 * Tt * 2);       // 4 MB
    float*    f    = (float*)   carve((size_t)Bb * NN * 4);       // 8 MB
    float*    fn   = (float*)   carve((size_t)Bb * NN * 4);       // 8 MB
    _Float16* W1t  = (_Float16*)carve((size_t)Dd * Tt * 2);
    _Float16* W2t  = (_Float16*)carve((size_t)Tt * Tt * 2);
    _Float16* W3t  = (_Float16*)carve((size_t)Tt * Tt * 2);
    float*    gpre = (float*)   carve((size_t)Bb * Tt * 4);
    (void)ws_size; (void)in_sizes; (void)n_in; (void)out_size;

    // weight prep (tiny)
    wprep_kernel<<<(Dd * Tt + 255) / 256, 256, 0, stream>>>(W1, W1t, Dd, Tt);
    wprep_kernel<<<(Tt * Tt + 255) / 256, 256, 0, stream>>>(W2, W2t, Tt, Tt);
    wprep_kernel<<<(Tt * Tt + 255) / 256, 256, 0, stream>>>(W3, W3t, Tt, Tt);

    // fused double-EMA scan
    ema_kernel<<<(Bb * Dd) / 256, 256, 0, stream>>>(x, e_h);

    // GEMM1: tanh(e @ W1 + b1)   [8192,512]x[512,256]
    gemm16_kernel<0><<<(M1 / 16) / 2, 256, 0, stream>>>(e_h, W1t, b1, h1, nullptr, M1, Tt, Dd);
    // GEMM2: relu(h1 @ W2 + b2)  [8192,256]x[256,256]
    gemm16_kernel<1><<<(M1 / 16) / 2, 256, 0, stream>>>(h1, W2t, b2, h2, nullptr, M1, Tt, Tt);
    // GEMM3: h2 @ W3 + b3 -> f (f32, already in [B, T*T] flatten order)
    gemm16_kernel<2><<<(M1 / 16) / 2, 256, 0, stream>>>(h2, W3t, b3, nullptr, f, M1, Tt, Tt);

    // LayerNorm over flattened T*T per batch
    ln_kernel<<<Bb, 256, 0, stream>>>(f, gamma, beta, fn);

    // g_pre = fn @ Wa (f32 WMMA, K-split + atomics)
    zero_kernel<<<(Bb * Tt + 255) / 256, 256, 0, stream>>>(gpre, Bb * Tt);
    gemmwa_kernel<<<128, 256, 0, stream>>>(fn, Wa, gpre);

    // out = sigmoid(g_pre + ba) * x
    finalize_kernel<<<(int)(((size_t)Bb * Tt * Dd) / 256), 256, 0, stream>>>(x, gpre, ba, out);
}
